// diffusionNetworks_60026462929246
// MI455X (gfx1250) — matrix operand order repaired and checked by reading, not compiled, not run on hardware
//
#include <hip/hip_runtime.h>

typedef __attribute__((ext_vector_type(16))) _Float16 v16h;
typedef __attribute__((ext_vector_type(8)))  float    v8f;

#define NNODES 50000
#define NEDGES 800000
#define NOPEN  64
#define NTILE_N (NNODES/16)   // 3125
#define NTILE_E (NEDGES/16)   // 50000
#define TPW_E 4               // edge tiles per wave (A-fragment reuse)
#define NWAVE_E (NTILE_E/TPW_E)  // 12500

__device__ __forceinline__ v8f wmma_f16(v16h a, v16h b, v8f c) {
  // D = A(16x32 f16) * B(32x16 f16) + C(16x16 f32)
  return __builtin_amdgcn_wmma_f32_16x16x32_f16(
      /*neg_a=*/false, a, /*neg_b=*/false, b,
      /*c_mod=*/(short)0, c, /*reuse_a=*/false, /*reuse_b=*/false);
}

// ---------------------------------------------------------------------------
// Pack a row-major weight matrix W (or W^T) into WMMA 16-bit A-fragment
// layout (contiguous k order, same as the B fragments): frag = [mtile][kchunk], 32 lanes x 16 halves each.
// lane L: m = mtile*16 + L%16, g = L/16
// elem e: k_local = 16g + e
// ---------------------------------------------------------------------------
__global__ void pack_a_kernel(const float* __restrict__ src, int M, int K,
                              int transpose, _Float16* __restrict__ dst,
                              int nMt, int nCk) {
  int tid  = blockIdx.x * blockDim.x + threadIdx.x;
  int frag = tid >> 5;
  int lane = tid & 31;
  if (frag >= nMt * nCk) return;
  int mt = frag / nCk, ck = frag % nCk;
  int g = lane >> 4, ml = lane & 15;
  int m = mt * 16 + ml;
  _Float16* o = dst + frag * 512 + lane * 16;
  for (int e = 0; e < 16; ++e) {
    int v = e >> 1, p = e & 1;
    int kl = 16 * g + e;
    int k = ck * 32 + kl;
    float val = 0.f;
    if (m < M && k < K) val = transpose ? src[k * M + m] : src[m * K + k];
    o[e] = (_Float16)val;
  }
}

__global__ void zero_kernel(float* __restrict__ p, int n) {
  int i = blockIdx.x * blockDim.x + threadIdx.x;
  if (i < n) p[i] = 0.f;
}

__device__ __forceinline__ void load_a8(const _Float16* __restrict__ apack,
                                        int fragBase, int lane, v16h a[8]) {
  const v16h* ap = (const v16h*)apack;
#pragma unroll
  for (int f = 0; f < 8; ++f) a[f] = ap[(fragBase + f) * 32 + lane];
}

// ---------------------------------------------------------------------------
// xn[64][N] = KNopen(64x16, K padded to 32 with zeros in frags) @ xn_in[16][N]
// One wave per 16-node tile, 4 WMMAs.
// ---------------------------------------------------------------------------
__global__ void __launch_bounds__(256) open_nodes_kernel(
    const float* __restrict__ xin, const _Float16* __restrict__ apack,
    float* __restrict__ xn) {
  int wid  = (blockIdx.x * blockDim.x + threadIdx.x) >> 5;  // wave-uniform
  int lane = threadIdx.x & 31;
  if (wid >= NTILE_N) return;
  int g = lane >> 4, n = lane & 15;
  int col = wid * 16 + n;
  v16h b;
#pragma unroll
  for (int t = 0; t < 16; ++t) b[t] = (_Float16)0.f;
  if (g == 0) {  // B rows k=0..15 (features); k=16..31 hit zero A entries
#pragma unroll
    for (int t = 0; t < 16; ++t) b[t] = (_Float16)xin[t * NNODES + col];
  }
  const v16h* ap = (const v16h*)apack;
#pragma unroll
  for (int mt = 0; mt < 4; ++mt) {
    v8f c = {};
    c = wmma_f16(ap[mt * 32 + lane], b, c);
#pragma unroll
    for (int r = 0; r < 8; ++r)
      xn[(mt * 16 + g * 8 + r) * NNODES + col] = c[r];
  }
}

// ---------------------------------------------------------------------------
// KNclose (3x64) @ xn -> XX slot (== coords for next layer); optionally also
// the xn output region (after last layer).
// ---------------------------------------------------------------------------
__global__ void close_kernel(const float* __restrict__ xn,
                             const float* __restrict__ KNclose,
                             float* __restrict__ xx, float* __restrict__ xnout) {
  int n = blockIdx.x * blockDim.x + threadIdx.x;
  if (n >= NNODES) return;
  float a0 = 0.f, a1 = 0.f, a2 = 0.f;
  for (int f = 0; f < 64; ++f) {
    float v = xn[f * NNODES + n];
    a0 += KNclose[f] * v;
    a1 += KNclose[64 + f] * v;
    a2 += KNclose[128 + f] * v;
  }
  xx[n] = a0; xx[NNODES + n] = a1; xx[2 * NNODES + n] = a2;
  if (xnout) { xnout[n] = a0; xnout[NNODES + n] = a1; xnout[2 * NNODES + n] = a2; }
}

// w[e] = exp(-10 * ||coords_i - coords_j||^2)
__global__ void wkern(const float* __restrict__ coords,
                      const int* __restrict__ iInd, const int* __restrict__ jInd,
                      float* __restrict__ wbuf) {
  int e = blockIdx.x * blockDim.x + threadIdx.x;
  if (e >= NEDGES) return;
  int i = iInd[e], j = jInd[e];
  float dx = coords[i] - coords[j];
  float dy = coords[NNODES + i] - coords[NNODES + j];
  float dz = coords[2 * NNODES + i] - coords[2 * NNODES + j];
  wbuf[e] = __expf(-10.f * (dx * dx + dy * dy + dz * dz));
}

// ---------------------------------------------------------------------------
// One tile: gather g = w*(xn[:,i]-xn[:,j]) for 16 edges, run
// Ai_tile = Kni(64x64) @ g via 8 WMMAs with preloaded A fragments.
// B layout: elem t -> k = 16g + t (chunk0) / 32 + 16g + t (chunk1).
// ---------------------------------------------------------------------------
__device__ __forceinline__ void edge_tile_mm(
    const float* __restrict__ xn, const float* __restrict__ wbuf,
    const int* __restrict__ iInd, const int* __restrict__ jInd,
    const v16h a[8], int tile, int lane, v8f acc[4]) {
  int g = lane >> 4, n = lane & 15;
  int e = tile * 16 + n;
  __builtin_prefetch(iInd + e + 16, 0, 0);  // global_prefetch next tile
  int i = iInd[e], j = jInd[e];
  float wv = wbuf[e];
  int r0 = g * 16, r1 = 32 + g * 16;
  v16h b0, b1;
#pragma unroll
  for (int t = 0; t < 16; ++t) {
    b0[t] = (_Float16)(wv * (xn[(r0 + t) * NNODES + i] - xn[(r0 + t) * NNODES + j]));
    b1[t] = (_Float16)(wv * (xn[(r1 + t) * NNODES + i] - xn[(r1 + t) * NNODES + j]));
  }
#pragma unroll
  for (int mt = 0; mt < 4; ++mt) {
    v8f c = {};
    c = wmma_f16(a[mt * 2 + 0], b0, c);
    c = wmma_f16(a[mt * 2 + 1], b1, c);
    acc[mt] = c;
  }
}

// Pass 1: instance-norm statistics. TPW_E tiles per wave, A fragments and the
// per-channel partial sums register-resident; one LDS-atomic flush per wave.
__global__ void __launch_bounds__(256) edge_stats_kernel(
    const float* __restrict__ xn, const float* __restrict__ wbuf,
    const int* __restrict__ iInd, const int* __restrict__ jInd,
    const _Float16* __restrict__ apack, int fragBase,
    float* __restrict__ stats) {
  __shared__ float s_sum[64], s_sq[64];
  if (threadIdx.x < 64) { s_sum[threadIdx.x] = 0.f; s_sq[threadIdx.x] = 0.f; }
  __syncthreads();
  int wid  = (blockIdx.x * blockDim.x + threadIdx.x) >> 5;  // wave-uniform
  int lane = threadIdx.x & 31;
  int g = lane >> 4;
  if (wid < NWAVE_E) {
    v16h a[8];
    load_a8(apack, fragBase, lane, a);
    v8f ssum[4], ssq[4];
#pragma unroll
    for (int mt = 0; mt < 4; ++mt) { v8f z = {}; ssum[mt] = z; ssq[mt] = z; }
    for (int t = 0; t < TPW_E; ++t) {
      v8f acc[4];
      edge_tile_mm(xn, wbuf, iInd, jInd, a, wid * TPW_E + t, lane, acc);
#pragma unroll
      for (int mt = 0; mt < 4; ++mt) {
        ssum[mt] += acc[mt];
        ssq[mt]  += acc[mt] * acc[mt];
      }
    }
#pragma unroll
    for (int mt = 0; mt < 4; ++mt)
#pragma unroll
      for (int r = 0; r < 8; ++r) {
        int m = mt * 16 + g * 8 + r;
        atomicAdd(&s_sum[m], ssum[mt][r]);
        atomicAdd(&s_sq[m],  ssq[mt][r]);
      }
  }
  __syncthreads();
  if (threadIdx.x < 64) {
    atomicAdd(&stats[threadIdx.x],      s_sum[threadIdx.x]);
    atomicAdd(&stats[64 + threadIdx.x], s_sq[threadIdx.x]);
  }
}

__global__ void finalize_stats(const float* __restrict__ stats,
                               float* __restrict__ nrm) {
  int m = threadIdx.x;
  if (m >= 64) return;
  float mean = stats[m] * (1.f / NEDGES);
  float var  = stats[64 + m] * (1.f / NEDGES) - mean * mean;
  nrm[m] = mean;
  nrm[64 + m] = rsqrtf(var + 1e-5f);
}

// ---------------------------------------------------------------------------
// Pass 2: recompute Ai, normalize+ReLU in registers (norm params from LDS),
// scatter w*xe into div (f32 atomics). Last layer: fuse xe_out = KEclose @ xe
// via LDS re-layout (D-fragment -> B-fragment) + 2 WMMAs per tile.
// ---------------------------------------------------------------------------
__global__ void __launch_bounds__(256) edge_scatter_kernel(
    const float* __restrict__ xn, const float* __restrict__ wbuf,
    const int* __restrict__ iInd, const int* __restrict__ jInd,
    const _Float16* __restrict__ apack, int fragBase,
    const float* __restrict__ nrm, float* __restrict__ divb,
    int doXe, int keFragBase, float* __restrict__ out_xe) {
  __shared__ _Float16 s_xe[8][64][16];
  __shared__ float s_nrm[128];
  if (threadIdx.x < 128) s_nrm[threadIdx.x] = nrm[threadIdx.x];
  __syncthreads();
  int wid  = (blockIdx.x * blockDim.x + threadIdx.x) >> 5;  // wave-uniform
  int lane = threadIdx.x & 31;
  int lwid = threadIdx.x >> 5;
  if (wid >= NWAVE_E) return;
  int g = lane >> 4, n = lane & 15;
  v16h a[8];
  load_a8(apack, fragBase, lane, a);
  const v16h* ap = (const v16h*)apack;
  v16h ka0 = ap[(keFragBase + 0) * 32 + lane];
  v16h ka1 = ap[(keFragBase + 1) * 32 + lane];
  for (int t = 0; t < TPW_E; ++t) {
    int tile = wid * TPW_E + t;
    int e = tile * 16 + n;
    int i = iInd[e], j = jInd[e];
    float wv = wbuf[e];
    v8f acc[4];
    edge_tile_mm(xn, wbuf, iInd, jInd, a, tile, lane, acc);
#pragma unroll
    for (int mt = 0; mt < 4; ++mt)
#pragma unroll
      for (int r = 0; r < 8; ++r) {
        int m = mt * 16 + g * 8 + r;
        float v = (acc[mt][r] - s_nrm[m]) * s_nrm[64 + m];
        v = fmaxf(v, 0.f);                 // xe = relu(instance_norm(Ai))
        float wg = wv * v;
        atomicAdd(&divb[m * NNODES + i],  wg);
        atomicAdd(&divb[m * NNODES + j], -wg);
        if (doXe) s_xe[lwid][m][n] = (_Float16)v;
      }
    if (doXe) {
      asm volatile("s_wait_dscnt 0" ::: "memory");  // wave-local LDS RAW
      v16h b0, b1;
#pragma unroll
      for (int q = 0; q < 16; ++q) {
        b0[q] = s_xe[lwid][g * 16 + q][n];
        b1[q] = s_xe[lwid][32 + g * 16 + q][n];
      }
      v8f c = {};
      c = wmma_f16(ka0, b0, c);
      c = wmma_f16(ka1, b1, c);
#pragma unroll
      for (int r = 0; r < 8; ++r)
        out_xe[(g * 8 + r) * NEDGES + e] = c[r];
    }
  }
}

// Bi = Kni^T @ div ; xn -= 0.1 * Bi. One wave per 16-node tile, 8 WMMAs.
__global__ void __launch_bounds__(256) node_update_kernel(
    const float* __restrict__ divb, const _Float16* __restrict__ apack,
    int fragBase, float* __restrict__ xn) {
  int wid  = (blockIdx.x * blockDim.x + threadIdx.x) >> 5;
  int lane = threadIdx.x & 31;
  if (wid >= NTILE_N) return;
  int g = lane >> 4, n = lane & 15;
  int col = wid * 16 + n;
  v16h a[8];
  load_a8(apack, fragBase, lane, a);
  v16h b0, b1;
#pragma unroll
  for (int t = 0; t < 16; ++t) {
    b0[t] = (_Float16)divb[(g * 16 + t) * NNODES + col];
    b1[t] = (_Float16)divb[(32 + g * 16 + t) * NNODES + col];
  }
#pragma unroll
  for (int mt = 0; mt < 4; ++mt) {
    v8f c = {};
    c = wmma_f16(a[mt * 2 + 0], b0, c);
    c = wmma_f16(a[mt * 2 + 1], b1, c);
#pragma unroll
    for (int r = 0; r < 8; ++r) {
      int m = mt * 16 + g * 8 + r;
      xn[m * NNODES + col] -= 0.1f * c[r];
    }
  }
}

// ---------------------------------------------------------------------------
extern "C" void kernel_launch(void* const* d_in, const int* in_sizes, int n_in,
                              void* d_out, int out_size, void* d_ws, size_t ws_size,
                              hipStream_t stream) {
  (void)in_sizes; (void)n_in; (void)out_size; (void)ws_size;
  const float* xn_in   = (const float*)d_in[0];
  // d_in[1] (xe) and d_in[5] (KEopen) are dead: loop overwrites xe = relu(Ai)
  const int*   iInd    = (const int*)d_in[2];
  const int*   jInd    = (const int*)d_in[3];
  const float* KNopen  = (const float*)d_in[4];
  const float* KNclose = (const float*)d_in[6];
  const float* KEclose = (const float*)d_in[7];
  const float* KN      = (const float*)d_in[8];

  float* xn    = (float*)d_ws;           // [64][NNODES]
  float* divb  = xn + 64 * NNODES;       // [64][NNODES]
  float* wbuf  = divb + 64 * NNODES;     // [NEDGES]
  float* stats = wbuf + NEDGES;          // sum[64], sumsq[64]
  float* nrm   = stats + 128;            // mean[64], invstd[64]
  _Float16* apack = (_Float16*)(nrm + 128);  // 54 frags * 512 halves

  float* out_xn = (float*)d_out;             // [3][NNODES]
  float* out_xe = out_xn + 3 * NNODES;       // [16][NEDGES]
  float* out_XX = out_xe + 16 * NEDGES;      // [4][3][NNODES]

  // --- pack weight fragments ---
  // KNopen: frags 0..3 ; KN[l]: 4+16l .. +7 ; KN[l]^T: 4+16l+8 .. +15 ; KEclose: 52,53
  pack_a_kernel<<<1, 128, 0, stream>>>(KNopen, 64, 16, 0, apack, 4, 1);
  for (int l = 0; l < 3; ++l) {
    pack_a_kernel<<<1, 256, 0, stream>>>(KN + l * 64 * 64, 64, 64, 0,
                                         apack + (4 + l * 16) * 512, 4, 2);
    pack_a_kernel<<<1, 256, 0, stream>>>(KN + l * 64 * 64, 64, 64, 1,
                                         apack + (4 + l * 16 + 8) * 512, 4, 2);
  }
  pack_a_kernel<<<1, 64, 0, stream>>>(KEclose, 16, 64, 0, apack + 52 * 512, 1, 2);

  // --- open + XX[0] (== coords for layer 0) ---
  open_nodes_kernel<<<(NTILE_N * 32 + 255) / 256, 256, 0, stream>>>(xn_in, apack, xn);
  close_kernel<<<(NNODES + 255) / 256, 256, 0, stream>>>(xn, KNclose, out_XX, nullptr);

  for (int l = 0; l < 3; ++l) {
    int knBase  = 4 + l * 16;
    int kntBase = knBase + 8;
    zero_kernel<<<(64 * NNODES + 255) / 256, 256, 0, stream>>>(divb, 64 * NNODES);
    zero_kernel<<<1, 256, 0, stream>>>(stats, 128);
    wkern<<<(NEDGES + 255) / 256, 256, 0, stream>>>(out_XX + l * 3 * NNODES,
                                                    iInd, jInd, wbuf);
    edge_stats_kernel<<<(NWAVE_E + 7) / 8, 256, 0, stream>>>(
        xn, wbuf, iInd, jInd, apack, knBase, stats);
    finalize_stats<<<1, 64, 0, stream>>>(stats, nrm);
    edge_scatter_kernel<<<(NWAVE_E + 7) / 8, 256, 0, stream>>>(
        xn, wbuf, iInd, jInd, apack, knBase, nrm, divb,
        /*doXe=*/(l == 2), /*keFragBase=*/52, out_xe);
    node_update_kernel<<<(NTILE_N + 7) / 8, 256, 0, stream>>>(divb, apack, kntBase, xn);
    close_kernel<<<(NNODES + 255) / 256, 256, 0, stream>>>(
        xn, KNclose, out_XX + (l + 1) * 3 * NNODES,
        (l == 2) ? out_xn : nullptr);  // xn_out == XX[3]
  }
}
